// DecoderCell_78838419685752
// MI455X (gfx1250) — matrix operand order repaired and checked by reading, not compiled, not run on hardware
//
#include <hip/hip_runtime.h>

#define BATCH 512
#define TLEN  512
#define ENCD  512
#define ATTND 256
#define DECD  256
#define NMELS 80
#define HIDD  128
#define DCC   8
#define DKK   21
#define SCC   8
#define SKK   21
#define PLL   11

typedef __attribute__((ext_vector_type(16))) __bf16 v16bf;
typedef __attribute__((ext_vector_type(8)))  float  v8f;

union BF16x16 { v16bf v; __bf16 e[16]; };
union F32x8   { v8f v; float f[8]; };

__device__ __forceinline__ __bf16 f2bf(float x) { return (__bf16)x; }

// Fast transcendentals: CDNA5 has native TANH/EXP/LOG trans-VALU ops.
__device__ __forceinline__ float fast_tanh(float x) {
#if __has_builtin(__builtin_amdgcn_tanhf)
  return __builtin_amdgcn_tanhf(x);
#else
  return tanhf(x);
#endif
}
__device__ __forceinline__ float sigmoidf(float x) { return 1.f / (1.f + __expf(-x)); }

// ---------------------------------------------------------------------------
// Y = act(X @ W^T + bias [+ addsrc]) GEMM, bf16 WMMA, fp32 accum.
// X:[M,K] ldx, W:[N,K] ldw (torch weight layout), Y:[M,N] ldy.
// One wave -> 16x64 macro-tile: A fragment loaded once per K-step, reused by
// 4 consecutive N sub-tiles (4 accumulators, 4 back-to-back WMMAs).
// NFULL=true  : N % 64 == 0 -> no column guards at all (branch-free hot loop).
// NFULL=false : per-sub-tile column guards for ragged N (168, 160).
// Branch-free main K loop; single guarded tail step. Requires M % 16 == 0.
// act: 0=none 1=relu 2=tanh.
// ---------------------------------------------------------------------------
template <bool NFULL>
__global__ void gemm_bf16_wmma(const float* __restrict__ X, int ldx,
                               const float* __restrict__ W, int ldw,
                               const float* __restrict__ bias,
                               const float* __restrict__ addsrc,
                               float* __restrict__ Y, int ldy,
                               int M, int N, int K, int act) {
  int nt64 = (N + 63) >> 6;
  int mt   = (M + 15) >> 4;
  int tile = (int)((blockIdx.x * blockDim.x + threadIdx.x) >> 5);
  if (tile >= nt64 * mt) return;
  int tm = tile / nt64, tn = tile - tm * nt64;
  int m0 = tm << 4, n0 = tn << 6;
  int lane = (int)(threadIdx.x & 31);
  int half = lane >> 4, l16 = lane & 15;
  int row = m0 + l16;
  size_t xbase = (size_t)row * ldx;

  F32x8 acc[4];
#pragma unroll
  for (int s = 0; s < 4; ++s)
#pragma unroll
    for (int r = 0; r < 8; ++r) acc[s].f[r] = 0.f;

  int kmain = K & ~31;

  // ---- main loop: no K guards; column guards only in the ragged-N variant ----
  for (int k0 = 0; k0 < kmain; k0 += 32) {
    BF16x16 a;
#pragma unroll
    for (int e = 0; e < 16; ++e) {
      int ka = k0 + (half << 4) + e;
      a.e[e] = f2bf(X[xbase + ka]);
    }
#pragma unroll
    for (int s = 0; s < 4; ++s) {
      int col = n0 + (s << 4) + l16;
      BF16x16 b;
      if (NFULL) {
#pragma unroll
        for (int e = 0; e < 16; ++e) {
          int kb = k0 + (half << 4) + e;
          b.e[e] = f2bf(W[(size_t)col * ldw + kb]);
        }
      } else {
#pragma unroll
        for (int e = 0; e < 16; ++e) {
          int kb = k0 + (half << 4) + e;
          b.e[e] = (col < N) ? f2bf(W[(size_t)col * ldw + kb]) : f2bf(0.f);
        }
      }
      acc[s].v = __builtin_amdgcn_wmma_f32_16x16x32_bf16(false, a.v, false, b.v,
                                                         (short)0, acc[s].v, false, false);
    }
  }

  // ---- K tail (K % 32 != 0): fully guarded ----
  if (kmain < K) {
    BF16x16 a;
#pragma unroll
    for (int e = 0; e < 16; ++e) {
      int ka = kmain + (half << 4) + e;
      a.e[e] = (ka < K) ? f2bf(X[xbase + ka]) : f2bf(0.f);
    }
#pragma unroll
    for (int s = 0; s < 4; ++s) {
      int col = n0 + (s << 4) + l16;
      BF16x16 b;
#pragma unroll
      for (int e = 0; e < 16; ++e) {
        int kb = kmain + (half << 4) + e;
        bool ok = NFULL ? (kb < K) : (col < N && kb < K);
        b.e[e] = ok ? f2bf(W[(size_t)col * ldw + kb]) : f2bf(0.f);
      }
      acc[s].v = __builtin_amdgcn_wmma_f32_16x16x32_bf16(false, a.v, false, b.v,
                                                         (short)0, acc[s].v, false, false);
    }
  }

  // ---- epilogue ----
#pragma unroll
  for (int s = 0; s < 4; ++s) {
    int n = n0 + (s << 4) + l16;
    if (NFULL || n < N) {
      float bi = bias ? bias[n] : 0.f;
#pragma unroll
      for (int r = 0; r < 8; ++r) {
        int m = m0 + (half << 3) + r;
        float v = acc[s].f[r] + bi;
        if (addsrc) v += addsrc[(size_t)m * ldy + n];
        if (act == 1)      v = fmaxf(v, 0.f);
        else if (act == 2) v = fast_tanh(v);
        Y[(size_t)m * ldy + n] = v;
      }
    }
  }
}

// dst[M, ka+kb] = concat(a[M,ka], b[M,kb])
__global__ void concat2_kernel(const float* __restrict__ a, int ka,
                               const float* __restrict__ b, int kb,
                               float* __restrict__ dst, int M) {
  int K = ka + kb;
  int idx = blockIdx.x * blockDim.x + threadIdx.x;
  if (idx >= M * K) return;
  int m = idx / K, k = idx - m * K;
  dst[idx] = (k < ka) ? a[(size_t)m * ka + k] : b[(size_t)m * kb + (k - ka)];
}

// LSTM pointwise: gates[b,4H] order (i,f,g,o); optional residual xres = x_in + h.
__global__ void lstm_pw_kernel(const float* __restrict__ gates,
                               const float* __restrict__ c_prev,
                               float* __restrict__ h_out, float* __restrict__ c_out,
                               const float* __restrict__ x_in, float* __restrict__ xres,
                               int H, int n) {
  int idx = blockIdx.x * blockDim.x + threadIdx.x;
  if (idx >= n) return;
  int b = idx / H, j = idx - b * H;
  const float* g = gates + (size_t)b * 4 * H;
  float i  = sigmoidf(g[j]);
  float f  = sigmoidf(g[H + j]);
  float gg = fast_tanh(g[2 * H + j]);
  float o  = sigmoidf(g[3 * H + j]);
  float cn = f * c_prev[idx] + i * gg;
  float hn = o * fast_tanh(cn);
  h_out[idx] = hn;
  c_out[idx] = cn;
  if (xres) xres[idx] = x_in[idx] + hn;
}

// Prior conv + static conv f + dynamic (per-batch) conv g.
// fg[row,0:8]=f, fg[row,8:16]=g ; p[row]=log(clip(prior)).
__global__ void dca_fgp_kernel(const float* __restrict__ alpha,
                               const float* __restrict__ P,
                               const float* __restrict__ Fw,     // [8,21]
                               const float* __restrict__ G,      // [B,8,21]
                               float* __restrict__ fg, float* __restrict__ p) {
  int idx = blockIdx.x * blockDim.x + threadIdx.x;
  if (idx >= BATCH * TLEN) return;
  int b = idx / TLEN, t = idx - b * TLEN;
  const float* ab = alpha + (size_t)b * TLEN;
  float pp = 0.f;
#pragma unroll
  for (int k = 0; k < PLL; ++k) {
    int tt = t + k - (PLL - 1);
    if (tt >= 0) pp += ab[tt] * P[k];
  }
  p[idx] = __logf(fmaxf(pp, 1e-6f));
  float w[DKK];
#pragma unroll
  for (int k = 0; k < DKK; ++k) {
    int tt = t + k - (DKK - 1) / 2;
    w[k] = (tt >= 0 && tt < TLEN) ? ab[tt] : 0.f;
  }
  const float* Gb = G + (size_t)b * DCC * DKK;
#pragma unroll
  for (int c = 0; c < DCC; ++c) {
    float fs = 0.f, gs = 0.f;
#pragma unroll
    for (int k = 0; k < DKK; ++k) {
      fs += w[k] * Fw[c * SKK + k];
      gs += w[k] * Gb[c * DKK + k];
    }
    fg[(size_t)idx * 16 + c]     = fs;
    fg[(size_t)idx * 16 + 8 + c] = gs;
  }
}

// Fused energy: e = tanh([f,g] @ [U;T]^T + T_b) @ v^T + p.
// One wave handles 16 (b,t) rows; K=16 (padded to 32), 8 WMMAs over N=128.
// B-fragment: every lane loads the Uw/Tw rows unconditionally (addresses are
// valid for all lanes) and the upper-half lanes are zeroed with a value
// select (v_cndmask) -> no exec-mask divergence in the hot loop.
__global__ void dca_e_kernel(const float* __restrict__ fg,
                             const float* __restrict__ Uw,  // [128,8]
                             const float* __restrict__ Tw,  // [128,8]
                             const float* __restrict__ Tb,  // [128]
                             const float* __restrict__ vw,  // [128]
                             const float* __restrict__ p,
                             float* __restrict__ e_out, int rows) {
  int wave = (int)((blockIdx.x * blockDim.x + threadIdx.x) >> 5);
  int m0 = wave << 4;
  if (m0 >= rows) return;
  int lane = (int)(threadIdx.x & 31);
  int half = lane >> 4, l16 = lane & 15;
  int row = m0 + l16;

  BF16x16 a;
#pragma unroll
  for (int e = 0; e < 16; ++e) {
    int k = (half << 4) + e;   // half 0 -> k<16, half 1 -> k>=16
    a.e[e] = (k < 16) ? f2bf(fg[(size_t)row * 16 + k]) : f2bf(0.f);
  }

  float partial[8];
#pragma unroll
  for (int r = 0; r < 8; ++r) partial[r] = 0.f;

  for (int nt = 0; nt < 8; ++nt) {
    int col = (nt << 4) + l16;          // < 128 for every lane
    const float* ur = Uw + col * 8;
    const float* tr = Tw + col * 8;
    BF16x16 b;
#pragma unroll
    for (int e = 0; e < 8; ++e) {
      float u = ur[e];                  // unconditional loads, all lanes
      float t = tr[e];
      b.e[e]     = f2bf(half ? 0.f : u);  // lanes 16-31 carry K>=16 -> zero
      b.e[e + 8] = f2bf(half ? 0.f : t);
    }
    F32x8 acc;
    float bias = Tb[col];
#pragma unroll
    for (int r = 0; r < 8; ++r) acc.f[r] = bias;
    acc.v = __builtin_amdgcn_wmma_f32_16x16x32_bf16(false, a.v, false, b.v,
                                                    (short)0, acc.v, false, false);
    float vv = vw[col];
#pragma unroll
    for (int r = 0; r < 8; ++r) partial[r] += fast_tanh(acc.f[r]) * vv;
  }

#pragma unroll
  for (int r = 0; r < 8; ++r) {
    float s = partial[r];
    s += __shfl_xor(s, 1);
    s += __shfl_xor(s, 2);
    s += __shfl_xor(s, 4);
    s += __shfl_xor(s, 8);
    partial[r] = s;
  }
  if (l16 == 0) {
#pragma unroll
    for (int r = 0; r < 8; ++r) {
      int m = m0 + (half << 3) + r;
      e_out[m] = partial[r] + p[m];
    }
  }
}

// Row softmax over T=512, one block (256 threads, 2 elems/thread) per batch row.
__global__ void softmax_kernel(const float* __restrict__ e, float* __restrict__ out) {
  __shared__ float sm[256];
  int b = blockIdx.x;
  int tid = threadIdx.x;
  float v0 = e[(size_t)b * TLEN + tid];
  float v1 = e[(size_t)b * TLEN + 256 + tid];
  sm[tid] = fmaxf(v0, v1);
  __syncthreads();
  for (int s = 128; s > 0; s >>= 1) {
    if (tid < s) sm[tid] = fmaxf(sm[tid], sm[tid + s]);
    __syncthreads();
  }
  float mx = sm[0];
  __syncthreads();
  float e0 = __expf(v0 - mx), e1 = __expf(v1 - mx);
  sm[tid] = e0 + e1;
  __syncthreads();
  for (int s = 128; s > 0; s >>= 1) {
    if (tid < s) sm[tid] += sm[tid + s];
    __syncthreads();
  }
  float inv = 1.f / sm[0];
  out[(size_t)b * TLEN + tid]       = e0 * inv;
  out[(size_t)b * TLEN + 256 + tid] = e1 * inv;
}

// c2[b,d] = sum_t alpha2[b,t] * h[b,t,d]. Streams 512 MB of h; BW-bound.
__global__ void context_kernel(const float* __restrict__ alpha2,
                               const float* __restrict__ h,
                               float* __restrict__ c2) {
  int b = blockIdx.x;
  int d = threadIdx.x;  // 256 threads * float2 = 512 floats
  const float2* hb = (const float2*)(h + (size_t)b * TLEN * ENCD);
  const float* ab = alpha2 + (size_t)b * TLEN;
  float2 acc; acc.x = 0.f; acc.y = 0.f;
#pragma unroll 4
  for (int t = 0; t < TLEN; ++t) {
    __builtin_prefetch(&hb[(size_t)(t + 16) * (ENCD / 2) + d], 0, 0);
    float s = ab[t];
    float2 v = hb[(size_t)t * (ENCD / 2) + d];
    acc.x += s * v.x;
    acc.y += s * v.y;
  }
  ((float2*)(c2 + (size_t)b * ENCD))[d] = acc;
}

// ---------------------------------------------------------------------------
static void gemm(hipStream_t s, const float* X, int ldx, const float* W, int ldw,
                 const float* bias, const float* add, float* Y, int ldy,
                 int M, int N, int K, int act) {
  int tiles = ((M + 15) / 16) * ((N + 63) / 64);   // 16x64 macro-tiles
  int blocks = (tiles + 7) / 8;                    // 8 waves (256 threads) per block
  if ((N & 63) == 0)
    gemm_bf16_wmma<true><<<blocks, 256, 0, s>>>(X, ldx, W, ldw, bias, add, Y, ldy, M, N, K, act);
  else
    gemm_bf16_wmma<false><<<blocks, 256, 0, s>>>(X, ldx, W, ldw, bias, add, Y, ldy, M, N, K, act);
}

extern "C" void kernel_launch(void* const* d_in, const int* in_sizes, int n_in,
                              void* d_out, int out_size, void* d_ws, size_t ws_size,
                              hipStream_t stream) {
  const float* h        = (const float*)d_in[0];
  const float* y        = (const float*)d_in[1];
  const float* alpha    = (const float*)d_in[2];
  const float* cin      = (const float*)d_in[3];
  const float* attn_h0  = (const float*)d_in[4];
  const float* attn_c0  = (const float*)d_in[5];
  const float* r1_h0    = (const float*)d_in[6];
  const float* r1_c0    = (const float*)d_in[7];
  const float* r2_h0    = (const float*)d_in[8];
  const float* r2_c0    = (const float*)d_in[9];
  const float* fc1_w    = (const float*)d_in[10];
  const float* fc1_b    = (const float*)d_in[11];
  const float* fc2_w    = (const float*)d_in[12];
  const float* fc2_b    = (const float*)d_in[13];
  const float* W_w      = (const float*)d_in[14];
  const float* W_b      = (const float*)d_in[15];
  const float* V_w      = (const float*)d_in[16];
  const float* F_w      = (const float*)d_in[17];
  const float* U_w      = (const float*)d_in[18];
  const float* T_w      = (const float*)d_in[19];
  const float* T_b      = (const float*)d_in[20];
  const float* v_w      = (const float*)d_in[21];
  const float* P        = (const float*)d_in[22];
  const float* attn_wih = (const float*)d_in[23];
  const float* attn_whh = (const float*)d_in[24];
  const float* attn_bih = (const float*)d_in[25];
  const float* attn_bhh = (const float*)d_in[26];
  const float* lin_w    = (const float*)d_in[27];
  const float* lin_b    = (const float*)d_in[28];
  const float* r1_wih   = (const float*)d_in[29];
  const float* r1_whh   = (const float*)d_in[30];
  const float* r1_bih   = (const float*)d_in[31];
  const float* r1_bhh   = (const float*)d_in[32];
  const float* r2_wih   = (const float*)d_in[33];
  const float* r2_whh   = (const float*)d_in[34];
  const float* r2_bih   = (const float*)d_in[35];
  const float* r2_bhh   = (const float*)d_in[36];
  const float* proj_w   = (const float*)d_in[37];

  // d_out layout (floats): out | alpha2 | c2 | attn_h | attn_c | r1_h | r1_c | r2_h | r2_c
  float* out      = (float*)d_out;                      // 512*160
  float* o_alpha2 = out + BATCH * NMELS * 2;            // 512*512
  float* o_c2     = o_alpha2 + BATCH * TLEN;            // 512*512
  float* o_ah     = o_c2 + BATCH * ENCD;                // 512*256
  float* o_ac     = o_ah + BATCH * ATTND;
  float* o_r1h    = o_ac + BATCH * ATTND;
  float* o_r1c    = o_r1h + BATCH * DECD;
  float* o_r2h    = o_r1c + BATCH * DECD;
  float* o_r2c    = o_r2h + BATCH * DECD;

  // workspace layout
  float* ws = (float*)d_ws;
  size_t o = 0;
  float* yp1   = ws + o; o += (size_t)BATCH * 256;
  float* ypb   = ws + o; o += (size_t)BATCH * 256;
  float* xcat  = ws + o; o += (size_t)BATCH * 768;
  float* gates = ws + o; o += (size_t)BATCH * 1024;
  float* Gt    = ws + o; o += (size_t)BATCH * HIDD;
  float* Gbuf  = ws + o; o += (size_t)BATCH * DCC * DKK;
  float* pbuf  = ws + o; o += (size_t)BATCH * TLEN;
  float* fg    = ws + o; o += (size_t)BATCH * TLEN * 16;
  float* ebuf  = ws + o; o += (size_t)BATCH * TLEN;
  float* xcat2 = ws + o; o += (size_t)BATCH * 768;
  float* xb    = ws + o; o += (size_t)BATCH * 256;
  float* x2b   = ws + o; o += (size_t)BATCH * 256;
  float* x3b   = ws + o; o += (size_t)BATCH * 256;
  (void)ws_size; (void)in_sizes; (void)n_in; (void)out_size;

  // --- prenet ---
  gemm(stream, y, NMELS, fc1_w, NMELS, fc1_b, nullptr, yp1, 256, BATCH, 256, NMELS, 1);
  gemm(stream, yp1, 256, fc2_w, 256, fc2_b, nullptr, ypb, 256, BATCH, 256, 256, 1);

  // --- attention LSTM ---
  {
    int n = BATCH * 768;
    concat2_kernel<<<(n + 255) / 256, 256, 0, stream>>>(cin, ENCD, ypb, 256, xcat, BATCH);
  }
  gemm(stream, xcat, 768, attn_wih, 768, attn_bih, nullptr, gates, 1024, BATCH, 1024, 768, 0);
  gemm(stream, attn_h0, 256, attn_whh, 256, attn_bhh, gates, gates, 1024, BATCH, 1024, 256, 0);
  lstm_pw_kernel<<<(BATCH * ATTND + 255) / 256, 256, 0, stream>>>(
      gates, attn_c0, o_ah, o_ac, nullptr, nullptr, ATTND, BATCH * ATTND);

  // --- dynamic convolution attention ---
  gemm(stream, o_ah, 256, W_w, 256, W_b, nullptr, Gt, HIDD, BATCH, HIDD, 256, 2);      // tanh
  gemm(stream, Gt, HIDD, V_w, HIDD, nullptr, nullptr, Gbuf, DCC * DKK, BATCH, DCC * DKK, HIDD, 0);
  dca_fgp_kernel<<<(BATCH * TLEN + 255) / 256, 256, 0, stream>>>(alpha, P, F_w, Gbuf, fg, pbuf);
  {
    int rows = BATCH * TLEN;               // 16 rows per wave, 8 waves per block
    int blocks = rows / (16 * 8);
    dca_e_kernel<<<blocks, 256, 0, stream>>>(fg, U_w, T_w, T_b, v_w, pbuf, ebuf, rows);
  }
  softmax_kernel<<<BATCH, 256, 0, stream>>>(ebuf, o_alpha2);

  // --- context (streams 512 MB of h; bandwidth-bound) ---
  context_kernel<<<BATCH, 256, 0, stream>>>(o_alpha2, h, o_c2);

  // --- decoder stack ---
  {
    int n = BATCH * 768;
    concat2_kernel<<<(n + 255) / 256, 256, 0, stream>>>(o_c2, ENCD, o_ah, ATTND, xcat2, BATCH);
  }
  gemm(stream, xcat2, 768, lin_w, 768, lin_b, nullptr, xb, 256, BATCH, 256, 768, 0);

  gemm(stream, xb, 256, r1_wih, 256, r1_bih, nullptr, gates, 1024, BATCH, 1024, 256, 0);
  gemm(stream, r1_h0, 256, r1_whh, 256, r1_bhh, gates, gates, 1024, BATCH, 1024, 256, 0);
  lstm_pw_kernel<<<(BATCH * DECD + 255) / 256, 256, 0, stream>>>(
      gates, r1_c0, o_r1h, o_r1c, xb, x2b, DECD, BATCH * DECD);

  gemm(stream, x2b, 256, r2_wih, 256, r2_bih, nullptr, gates, 1024, BATCH, 1024, 256, 0);
  gemm(stream, r2_h0, 256, r2_whh, 256, r2_bhh, gates, gates, 1024, BATCH, 1024, 256, 0);
  lstm_pw_kernel<<<(BATCH * DECD + 255) / 256, 256, 0, stream>>>(
      gates, r2_c0, o_r2h, o_r2c, x2b, x3b, DECD, BATCH * DECD);

  // --- projection (writes [512,160] == [512,80,2] flat) ---
  gemm(stream, x3b, 256, proj_w, 256, nullptr, nullptr, out, NMELS * 2, BATCH, NMELS * 2, 256, 0);
}
